// SeqSeqModel_25065429139757
// MI455X (gfx1250) — compile-verified, matrix operand order checked by reading, not run on hardware
//
#include <hip/hip_runtime.h>
#include <hip/hip_bf16.h>
#include <stdint.h>

// Problem constants (from reference)
#define PB   16
#define PT   128
#define PN   1024
#define PH   128
#define PIN  32
#define PE   32768

#define ROWS_TOTAL   (PB * PN)          // 16384 rows of state
#define STATE_ELEMS  (ROWS_TOTAL * PH)  // 2,097,152 f32
#define OUT_ELEMS    (PB * PT * PN)     // 2,097,152 (loss) ; outputs same

#define WMSG_E (PH * PH)          // 16384
#define WMIX_E (PH * (PH + PIN))  // 20480
#define WIH_E  (3 * PH * PH)      // 49152
#define WHH_E  (3 * PH * PH)      // 49152

typedef __attribute__((ext_vector_type(16))) __bf16 bf16x16;
typedef __attribute__((ext_vector_type(8)))  float  f32x8;
typedef __attribute__((ext_vector_type(4)))  unsigned int u32x4;
typedef __attribute__((ext_vector_type(8)))  int  i32x8;
typedef __attribute__((ext_vector_type(4)))  int  i32x4;

#if defined(__has_builtin)
#if __has_builtin(__builtin_amdgcn_tensor_load_to_lds) && \
    __has_builtin(__builtin_amdgcn_s_wait_tensorcnt)
#define HAVE_TDM 1
#endif
#endif

// ---------- small helpers ----------
__device__ __forceinline__ __bf16 f2bf(float f) {
  unsigned u = __builtin_bit_cast(unsigned, f);
  unsigned r = u + 0x7FFFu + ((u >> 16) & 1u);      // round-to-nearest-even
  unsigned short h = (unsigned short)(r >> 16);
  return __builtin_bit_cast(__bf16, h);
}
__device__ __forceinline__ float bf2f(__bf16 b) {
  unsigned u = ((unsigned)__builtin_bit_cast(unsigned short, b)) << 16;
  return __builtin_bit_cast(float, u);
}
__device__ __forceinline__ float sigmoidf(float x) {
  return 1.0f / (1.0f + __expf(-x));
}

// ISA 7.12.2: 16-bit A-matrix 16x32 fragment. lane&15 = M row, lane>>4 = K half.
// VGPR p in 0..3 holds K = kh*8 + 2p + c ; p in 4..7 holds K = 16 + kh*8 + 2(p-4) + c.
__device__ __forceinline__ bf16x16 load_frag(const __bf16* __restrict__ base,
                                             int rowBase, int stride, int kBase,
                                             int lane) {
  const int r  = rowBase + (lane & 15);
  const int kh = (lane >> 4) << 3;  // 0 or 8
  const __bf16* __restrict__ p = base + r * stride + kBase;
  bf16x16 v;
#pragma unroll
  for (int e = 0; e < 16; ++e) {
    const int pp = e >> 1, c = e & 1;
    const int k  = (pp < 4) ? (kh + 2 * pp + c) : (16 + kh + 2 * (pp - 4) + c);
    v[e] = p[k];
  }
  return v;
}

__device__ __forceinline__ f32x8 wmma_bf16(bf16x16 a, bf16x16 b, f32x8 c) {
  // v_wmma_f32_16x16x32_bf16 : D = A(16x32) * B(32x16) + C(16x16)
  return __builtin_amdgcn_wmma_f32_16x16x32_bf16(false, a, false, b,
                                                 (short)0, c, false, false);
}

#ifdef HAVE_TDM
// Issue a TDM DMA of a rows x cols bf16 row-major tile (global -> LDS).
// D# built per CDNA5 ISA sec 8.3/8.4: group0 = {count, lds_addr, global_addr, type=2},
// group1 = {data_size=2B, tensor dims, tile dims, stride}, groups 2/3 = 0 (2-D tile).
__device__ __forceinline__ void tdm_load_2d_bf16(const __bf16* gsrc,
                                                 __bf16* lds_dst,
                                                 unsigned rows, unsigned cols) {
  unsigned lds_off = (unsigned)(size_t)lds_dst;        // LDS aperture low 32 bits
  unsigned long long ga = (unsigned long long)(size_t)gsrc;
  u32x4 g0;
  g0[0] = 1u;                                          // count=1 (valid user D#)
  g0[1] = lds_off;                                     // lds_addr
  g0[2] = (unsigned)ga;                                // global_addr[31:0]
  g0[3] = (unsigned)((ga >> 32) & 0x1FFFFFFu) | (2u << 30);  // addr[56:32] | type=2
  i32x8 g1;
  g1[0] = (int)(1u << 16);                             // data_size=1 (2 bytes)
  g1[1] = (int)(cols << 16);                           // tensor_dim0 low16
  g1[2] = (int)(rows << 16);                           // dim0 hi=0 | tensor_dim1 low16
  g1[3] = (int)(cols << 16);                           // dim1 hi=0 | tile_dim0
  g1[4] = (int)rows;                                   // tile_dim1 | tile_dim2=0
  g1[5] = (int)cols;                                   // tensor_dim0_stride low32
  g1[6] = 0;
  g1[7] = 0;
  i32x4 z4 = {0, 0, 0, 0};
#if __clang_major__ >= 23
  i32x8 z8 = {0, 0, 0, 0, 0, 0, 0, 0};
  __builtin_amdgcn_tensor_load_to_lds(g0, g1, z4, z4, z8, 0);
#else
  __builtin_amdgcn_tensor_load_to_lds(g0, g1, z4, z4, 0);
#endif
}
#endif

// ---------- K0: convert weights to bf16, zero state ----------
__global__ void k_prep(const float* __restrict__ Wmsg, const float* __restrict__ Wmix,
                       const float* __restrict__ Wih,  const float* __restrict__ Whh,
                       __bf16* __restrict__ wmsg, __bf16* __restrict__ wmix,
                       __bf16* __restrict__ wih,  __bf16* __restrict__ whh,
                       float* __restrict__ state) {
  int i = blockIdx.x * 256 + threadIdx.x;
  if (i < STATE_ELEMS) state[i] = 0.0f;
  if (i < WMSG_E) {
    wmsg[i] = f2bf(Wmsg[i]);
  } else if (i < WMSG_E + WMIX_E) {
    int j = i - WMSG_E;            wmix[j] = f2bf(Wmix[j]);
  } else if (i < WMSG_E + WMIX_E + WIH_E) {
    int j = i - WMSG_E - WMIX_E;   wih[j]  = f2bf(Wih[j]);
  } else if (i < WMSG_E + WMIX_E + WIH_E + WHH_E) {
    int j = i - WMSG_E - WMIX_E - WIH_E; whh[j] = f2bf(Whh[j]);
  }
}

// ---------- K1: m = state @ W_msg^T + b_msg ; also zero agg ----------
// Block: 256 thr (8 waves) -> 16 rows x 128 cols; wave w owns cols [16w,16w+16).
// W_msg (32KB bf16) is DMA'd into LDS by the Tensor Data Mover while the
// activation tile is staged by VALU; B-fragments then come from LDS.
__global__ void k_msg(const float* __restrict__ state, const __bf16* __restrict__ Wb,
                      const float* __restrict__ bias, float* __restrict__ m,
                      float* __restrict__ agg) {
  __shared__ __bf16 sA[16 * PH];   // 4 KB activations
  __shared__ __bf16 sW[PH * PH];   // 32 KB W_msg staged by TDM
  const int tid  = threadIdx.x;
  const int lane = tid & 31;
  const int wv   = tid >> 5;
  const int R0   = blockIdx.x * 16;

#ifdef HAVE_TDM
  if (tid < 32) {                      // one wave issues the DMA descriptor
    tdm_load_2d_bf16(Wb, sW, PH, PH);
  }
#endif

  for (int i = tid; i < 16 * PH; i += 256) {
    int r = i >> 7, h = i & 127;
    sA[i] = f2bf(state[(size_t)(R0 + r) * PH + h]);
  }

#ifdef HAVE_TDM
  if (tid < 32) __builtin_amdgcn_s_wait_tensorcnt(0);
#else
  for (int i = tid; i < PH * PH; i += 256) sW[i] = Wb[i];
#endif
  __syncthreads();

  const int colBase = wv * 16;
  f32x8 acc = {0.f, 0.f, 0.f, 0.f, 0.f, 0.f, 0.f, 0.f};
#pragma unroll
  for (int kb = 0; kb < PH; kb += 32) {
    bf16x16 a = load_frag(sA, 0, PH, kb, lane);
    bf16x16 b = load_frag(sW, colBase, PH, kb, lane);   // B[k][n] = W[n][k]
    acc = wmma_bf16(a, b, acc);
  }
  const int nn = colBase + (lane & 15);
  const float bv = bias[nn];
#pragma unroll
  for (int r = 0; r < 8; ++r) {
    int row = R0 + ((lane >> 4) << 3) + r;
    m[(size_t)row * PH + nn]   = acc[r] + bv;
    agg[(size_t)row * PH + nn] = 0.0f;
  }
}

// ---------- K2: agg[:, tgt] += m[:, src] (scatter over edges) ----------
// One wave per (batch, edge); 32 lanes x 4 f32 cover H=128.
__global__ void k_scatter(const float* __restrict__ m, float* __restrict__ agg,
                          const int* __restrict__ ei) {
  const int gw   = blockIdx.x * 8 + (threadIdx.x >> 5);  // 0 .. B*E-1
  const int lane = threadIdx.x & 31;
  const int b = gw >> 15;            // / E (E = 2^15)
  const int e = gw & (PE - 1);
  const int src = ei[e];
  const int tgt = ei[PE + e];
  const float* __restrict__ ms = m   + (size_t)(b * PN + src) * PH;
  float* __restrict__       ad = agg + (size_t)(b * PN + tgt) * PH;
#pragma unroll
  for (int j = 0; j < 4; ++j) {
    int h = lane + j * 32;
    atomicAdd(&ad[h], ms[h]);
  }
}

// ---------- K3: fused mix GEMM -> gate GEMMs -> GRU -> readout ----------
// Block: 256 thr (8 waves) handles 16 rows. 3 chained WMMA GEMMs via LDS.
__global__ void k_fused(int t, const float* __restrict__ x,
                        float* __restrict__ state, const float* __restrict__ agg,
                        const __bf16* __restrict__ Wmix, const __bf16* __restrict__ Wih,
                        const __bf16* __restrict__ Whh,
                        const float* __restrict__ bmix, const float* __restrict__ bih,
                        const float* __restrict__ bhh,
                        const float* __restrict__ wout, const float* __restrict__ bout,
                        float* __restrict__ outputs) {
  __shared__ __bf16 s_ax[16 * (PH + PIN)];  // [agg | x_t] bf16, 5 KB
  __shared__ __bf16 s_st[16 * PH];          // state bf16 (later reused for new_state), 4 KB
  __shared__ __bf16 s_m2[16 * PH];          // mix result bf16, 4 KB
  __shared__ float  s_gi[16 * 3 * PH];      // 24 KB
  __shared__ float  s_gh[16 * 3 * PH];      // 24 KB

  const int tid  = threadIdx.x;
  const int lane = tid & 31;
  const int wv   = tid >> 5;
  const int R0   = blockIdx.x * 16;         // global row = b*N + n
  const int b    = R0 >> 10;                // / N (N = 2^10)
  const int n0   = R0 & (PN - 1);

  // Pre-touch the big gate weights (L2-resident) -> global_prefetch_b8
  if (tid < 64) {
    __builtin_prefetch(Wih + tid * 768, 0, 1);
    __builtin_prefetch(Whh + tid * 768, 0, 1);
  }

  // Stage [agg | x_t] and state into LDS as bf16
  for (int i = tid; i < 16 * PH; i += 256) {
    int r = i >> 7, h = i & 127;
    s_ax[r * (PH + PIN) + h] = f2bf(agg[(size_t)(R0 + r) * PH + h]);
    s_st[i] = f2bf(state[(size_t)(R0 + r) * PH + h]);
  }
  for (int i = tid; i < 16 * PIN; i += 256) {
    int r = i >> 5, c = i & 31;
    s_ax[r * (PH + PIN) + PH + c] =
        f2bf(x[(((size_t)b * PT + t) * PN + (n0 + r)) * PIN + c]);
  }
  __syncthreads();

  // GEMM 1: m2 = [agg,x] @ Wmix^T + bmix   (K = 160, 5 chunks)
  {
    const int colBase = wv * 16;
    f32x8 acc = {0.f, 0.f, 0.f, 0.f, 0.f, 0.f, 0.f, 0.f};
#pragma unroll
    for (int kb = 0; kb < PH + PIN; kb += 32) {
      bf16x16 a  = load_frag(s_ax, 0, PH + PIN, kb, lane);
      bf16x16 bb = load_frag(Wmix, colBase, PH + PIN, kb, lane);
      acc = wmma_bf16(a, bb, acc);
    }
    const int nn = colBase + (lane & 15);
    const float bv = bmix[nn];
#pragma unroll
    for (int r = 0; r < 8; ++r) {
      int row = ((lane >> 4) << 3) + r;
      s_m2[row * PH + nn] = f2bf(acc[r] + bv);
    }
  }
  __syncthreads();

  // GEMM 2+3: gi = m2 @ Wih^T + bih ; gh = state @ Whh^T + bhh  (384 cols -> 3 tiles/wave)
#pragma unroll
  for (int tt = 0; tt < 3; ++tt) {
    const int colBase = tt * PH + wv * 16;
    f32x8 ai = {0.f, 0.f, 0.f, 0.f, 0.f, 0.f, 0.f, 0.f};
    f32x8 ah = {0.f, 0.f, 0.f, 0.f, 0.f, 0.f, 0.f, 0.f};
#pragma unroll
    for (int kb = 0; kb < PH; kb += 32) {
      bf16x16 am = load_frag(s_m2, 0, PH, kb, lane);
      bf16x16 bi = load_frag(Wih, colBase, PH, kb, lane);
      ai = wmma_bf16(am, bi, ai);
      bf16x16 as = load_frag(s_st, 0, PH, kb, lane);
      bf16x16 bh = load_frag(Whh, colBase, PH, kb, lane);
      ah = wmma_bf16(as, bh, ah);
    }
    const int nn = colBase + (lane & 15);
    const float bvi = bih[nn], bvh = bhh[nn];
#pragma unroll
    for (int r = 0; r < 8; ++r) {
      int row = ((lane >> 4) << 3) + r;
      s_gi[row * (3 * PH) + nn] = ai[r] + bvi;
      s_gh[row * (3 * PH) + nn] = ah[r] + bvh;
    }
  }
  __syncthreads();

  // GRU elementwise + state update; stash new_state (bf16) in s_st for readout
  for (int i = tid; i < 16 * PH; i += 256) {
    int r = i >> 7, h = i & 127;
    const float* gi = &s_gi[r * (3 * PH)];
    const float* gh = &s_gh[r * (3 * PH)];
    float rg = sigmoidf(gi[h]        + gh[h]);
    float zg = sigmoidf(gi[h + PH]   + gh[h + PH]);
    float ng = tanhf   (gi[h + 2*PH] + rg * gh[h + 2*PH]);
    float st = state[(size_t)(R0 + r) * PH + h];
    float ns = (1.0f - zg) * ng + zg * st;
    state[(size_t)(R0 + r) * PH + h] = ns;
    s_st[i] = f2bf(ns);
  }
  __syncthreads();

  // Readout: out[r] = new_state[r] . w_out + b_out
  if (tid < 16) {
    float s = 0.0f;
#pragma unroll 4
    for (int h = 0; h < PH; ++h) s += bf2f(s_st[tid * PH + h]) * wout[h];
    outputs[((size_t)b * PT + t) * PN + (n0 + tid)] = s + bout[0];
  }
}

// ---------- K4: loss = mask ? (out - target)^2 : 0 ----------
__global__ void k_loss(const float* __restrict__ outp, const float* __restrict__ tgt,
                       const uint8_t* __restrict__ mask, float* __restrict__ loss) {
  int i = blockIdx.x * 256 + threadIdx.x;
  if (i < OUT_ELEMS) {
    float d = outp[i] - tgt[i];
    loss[i] = mask[i] ? (d * d) : 0.0f;
  }
}

extern "C" void kernel_launch(void* const* d_in, const int* in_sizes, int n_in,
                              void* d_out, int out_size, void* d_ws, size_t ws_size,
                              hipStream_t stream) {
  const float* x       = (const float*)d_in[0];
  const float* targets = (const float*)d_in[1];
  const float* Wmsg    = (const float*)d_in[2];
  const float* bmsg    = (const float*)d_in[3];
  const float* Wmix    = (const float*)d_in[4];
  const float* bmix    = (const float*)d_in[5];
  const float* Wih     = (const float*)d_in[6];
  const float* bih     = (const float*)d_in[7];
  const float* Whh     = (const float*)d_in[8];
  const float* bhh     = (const float*)d_in[9];
  const float* Wout    = (const float*)d_in[10];
  const float* bout    = (const float*)d_in[11];
  const uint8_t* mask  = (const uint8_t*)d_in[12];
  const int*   ei      = (const int*)d_in[13];

  float* loss    = (float*)d_out;             // [B,T,N,1]
  float* outputs = loss + OUT_ELEMS;          // [B,T,N,1]

  // Workspace layout
  float*  st   = (float*)d_ws;                       // 8 MB
  float*  m    = st  + STATE_ELEMS;                  // 8 MB
  float*  agg  = m   + STATE_ELEMS;                  // 8 MB
  __bf16* wmsg = (__bf16*)(agg + STATE_ELEMS);
  __bf16* wmix = wmsg + WMSG_E;
  __bf16* wih  = wmix + WMIX_E;
  __bf16* whh  = wih  + WIH_E;

  k_prep<<<STATE_ELEMS / 256, 256, 0, stream>>>(Wmsg, Wmix, Wih, Whh,
                                                wmsg, wmix, wih, whh, st);

  for (int t = 0; t < PT; ++t) {
    k_msg<<<ROWS_TOTAL / 16, 256, 0, stream>>>(st, wmsg, bmsg, m, agg);
    k_scatter<<<(PB * PE) / 8, 256, 0, stream>>>(m, agg, ei);
    k_fused<<<ROWS_TOTAL / 16, 256, 0, stream>>>(t, x, st, agg, wmix, wih, whh,
                                                 bmix, bih, bhh, Wout, bout,
                                                 outputs);
  }

  k_loss<<<OUT_ELEMS / 256, 256, 0, stream>>>(outputs, targets, mask, loss);
}
